// TransformerEncoder_80831284510813
// MI455X (gfx1250) — matrix operand order repaired and checked by reading, not compiled, not run on hardware
//
#include <hip/hip_runtime.h>

#define USE_ASYNC_COPY 1

typedef __attribute__((ext_vector_type(16))) __bf16 v16bf;
typedef __attribute__((ext_vector_type(8)))  float  v8f;

__device__ __forceinline__ unsigned short f2bf(float f) {
  union { float f; unsigned u; } c; c.f = f;
  unsigned u = c.u;
  u += 0x7FFFu + ((u >> 16) & 1u);   // round-to-nearest-even
  return (unsigned short)(u >> 16);
}

union FragBF { v16bf v; uint4 q[2]; };

// ---------------------------------------------------------------------------
// fp32 -> bf16 elementwise convert (vectorized x4)
// ---------------------------------------------------------------------------
__global__ void convert_f32_bf16(const float* __restrict__ in,
                                 unsigned short* __restrict__ out, long n) {
  long i4 = ((long)blockIdx.x * blockDim.x + threadIdx.x) * 4;
  if (i4 < n) {
    float4 f = *(const float4*)(in + i4);
    union { unsigned short s[4]; uint2 q; } u;
    u.s[0] = f2bf(f.x); u.s[1] = f2bf(f.y); u.s[2] = f2bf(f.z); u.s[3] = f2bf(f.w);
    *(uint2*)(out + i4) = u.q;
  }
}

// ---------------------------------------------------------------------------
// fp32 [z][R][C] -> bf16 [z][C][R] transpose (weights -> [N,K] layout)
// ---------------------------------------------------------------------------
__global__ __launch_bounds__(256)
void transpose_f32_bf16(const float* __restrict__ in,
                        unsigned short* __restrict__ out, int R, int C) {
  long base = (long)blockIdx.z * (long)R * (long)C;
  __shared__ float tile[32][33];
  int c0 = blockIdx.x * 32, r0 = blockIdx.y * 32;
  int tx = threadIdx.x, ty = threadIdx.y;   // block (32,8)
  #pragma unroll
  for (int i = 0; i < 32; i += 8)
    tile[ty + i][tx] = in[base + (long)(r0 + ty + i) * C + c0 + tx];
  __syncthreads();
  #pragma unroll
  for (int i = 0; i < 32; i += 8)
    out[base + (long)(c0 + ty + i) * R + r0 + tx] = f2bf(tile[tx][ty + i]);
}

// ---------------------------------------------------------------------------
// WMMA bf16 GEMM: C[M,N] = scale * (A[M,K] x Bt[N,K]^T) + bias
//   block tile 128x128, BK=32, 8 waves (256 thr), wave tile 32x64
//   Double-buffered LDS; next tile fetched with async direct-to-LDS copies
//   (ASYNCcnt + s_wait_asynccnt), one barrier per K-step.
//   MODE 0: f32 store        MODE 1: bf16 store (opt. ReLU)
//   MODE 2: bf16 transposed store  out[b][n][m%rowsPerBatch]
//   batch z: offset = (z/zDiv)*Hi + (z%zDiv)*Lo per tensor (elements)
// ---------------------------------------------------------------------------
template<int MODE, bool RELU>
__global__ __launch_bounds__(256)
void gemm_wmma(const unsigned short* __restrict__ A,
               const unsigned short* __restrict__ Bt,
               void* __restrict__ Cv, const float* __restrict__ bias,
               int M, int N, int K, int lda, int ldb, int ldc, int zDiv,
               long aHi, long aLo, long bHi, long bLo,
               long cHi, long cLo, long biasHi, long biasLo,
               float scale, int rowsPerBatch) {
  (void)M;
  int z  = blockIdx.z;
  long zh = z / zDiv, zl = z % zDiv;
  A  += zh * aHi + zl * aLo;
  Bt += zh * bHi + zl * bLo;
  long cOff = zh * cHi + zl * cLo;
  const float* bptr = bias ? (bias + zh * biasHi + zl * biasLo) : nullptr;

  __shared__ __align__(16) unsigned short sA[2][128 * 40];  // 80B padded rows
  __shared__ __align__(16) unsigned short sB[2][128 * 40];
  const unsigned bufStride = 128 * 40 * 2;                  // bytes per buffer

  int tid  = threadIdx.x;
  int lane = tid & 31;
  int wid  = tid >> 5;       // 0..7
  int wm   = wid >> 1;       // 0..3 -> 32-row strip
  int wn   = wid & 1;        // 0..1 -> 64-col strip
  int sub  = lane & 15;
  int kh   = lane >> 4;      // lane half

  int m0 = blockIdx.y * 128;
  int n0 = blockIdx.x * 128;

  int lrow = tid >> 1;            // 0..127
  int lcol = (tid & 1) * 16;      // 0 or 16 (bf16 elems)

  const unsigned short* gaBase = A  + (long)(m0 + lrow) * lda + lcol;
  const unsigned short* gbBase = Bt + (long)(n0 + lrow) * ldb + lcol;
  // low 32 bits of the generic pointer = LDS byte address
  unsigned ldsA0 = (unsigned)(size_t)&sA[0][lrow * 40 + lcol];
  unsigned ldsB0 = (unsigned)(size_t)&sB[0][lrow * 40 + lcol];

  const v8f vzero = {0.f,0.f,0.f,0.f,0.f,0.f,0.f,0.f};
  v8f acc[2][4];
  #pragma unroll
  for (int i = 0; i < 2; i++)
    #pragma unroll
    for (int j = 0; j < 4; j++) acc[i][j] = vzero;

  int nt = K / 32;

#if USE_ASYNC_COPY
  // prologue: tile 0 -> buffer 0 via async DMA (each thread: 2 x 16B, offset
  // applies to both LDS and global address -> single base, offset:16)
  asm volatile(
      "global_load_async_to_lds_b128 %0, %2, off\n\t"
      "global_load_async_to_lds_b128 %0, %2, off offset:16\n\t"
      "global_load_async_to_lds_b128 %1, %3, off\n\t"
      "global_load_async_to_lds_b128 %1, %3, off offset:16"
      :: "v"(ldsA0), "v"(ldsB0), "v"(gaBase), "v"(gbBase) : "memory");
  asm volatile("s_wait_asynccnt 0" ::: "memory");
  __syncthreads();

  for (int it = 0; it < nt; ++it) {
    int cur = it & 1;
    int nxt = cur ^ 1;
    bool have = (it + 1) < nt;
    if (have) {
      const unsigned short* ga = gaBase + (it + 1) * 32;
      const unsigned short* gb = gbBase + (it + 1) * 32;
      unsigned la = ldsA0 + (unsigned)nxt * bufStride;
      unsigned lb = ldsB0 + (unsigned)nxt * bufStride;
      asm volatile(
          "global_load_async_to_lds_b128 %0, %2, off\n\t"
          "global_load_async_to_lds_b128 %0, %2, off offset:16\n\t"
          "global_load_async_to_lds_b128 %1, %3, off\n\t"
          "global_load_async_to_lds_b128 %1, %3, off offset:16"
          :: "v"(la), "v"(lb), "v"(ga), "v"(gb) : "memory");
      if (it + 2 < nt) __builtin_prefetch(ga + 32, 0, 3);
    }

    FragBF af[2], bf4[4];
    #pragma unroll
    for (int i = 0; i < 2; i++) {
      // A 16x32: elems 0..7 -> K=kh*8+0..7 ; elems 8..15 -> K=16+kh*8+0..7
      const unsigned short* p = &sA[cur][(wm * 32 + i * 16 + sub) * 40 + kh * 8];
      af[i].q[0] = *(const uint4*)p;
      af[i].q[1] = *(const uint4*)(p + 16);
    }
    #pragma unroll
    for (int j = 0; j < 4; j++) {
      // B 32x16 (as [N,K]): elems 0..7 -> K=kh*8+0..7 ; elems 8..15 -> K=16+kh*8+0..7
      const unsigned short* p = &sB[cur][(wn * 64 + j * 16 + sub) * 40 + kh * 8];
      bf4[j].q[0] = *(const uint4*)p;
      bf4[j].q[1] = *(const uint4*)(p + 16);
    }
    #pragma unroll
    for (int i = 0; i < 2; i++)
      #pragma unroll
      for (int j = 0; j < 4; j++)
        acc[i][j] = __builtin_amdgcn_wmma_f32_16x16x32_bf16(
            false, af[i].v, false, bf4[j].v, (short)0, acc[i][j], false, false);

    if (have) asm volatile("s_wait_asynccnt 0" ::: "memory");
    __syncthreads();
  }
#else
  // fallback: register-staged double buffering, one barrier per K-step
  {
    uint4 a0 = *(const uint4*)gaBase;
    uint4 a1 = *(const uint4*)(gaBase + 8);
    uint4 b0 = *(const uint4*)gbBase;
    uint4 b1 = *(const uint4*)(gbBase + 8);
    *(uint4*)&sA[0][lrow * 40 + lcol]     = a0;
    *(uint4*)&sA[0][lrow * 40 + lcol + 8] = a1;
    *(uint4*)&sB[0][lrow * 40 + lcol]     = b0;
    *(uint4*)&sB[0][lrow * 40 + lcol + 8] = b1;
  }
  __syncthreads();
  for (int it = 0; it < nt; ++it) {
    int cur = it & 1;
    int nxt = cur ^ 1;
    bool have = (it + 1) < nt;
    uint4 a0, a1, b0, b1;
    if (have) {
      const unsigned short* ga = gaBase + (it + 1) * 32;
      const unsigned short* gb = gbBase + (it + 1) * 32;
      a0 = *(const uint4*)ga;  a1 = *(const uint4*)(ga + 8);
      b0 = *(const uint4*)gb;  b1 = *(const uint4*)(gb + 8);
      if (it + 2 < nt) __builtin_prefetch(ga + 32, 0, 3);
    }
    FragBF af[2], bf4[4];
    #pragma unroll
    for (int i = 0; i < 2; i++) {
      const unsigned short* p = &sA[cur][(wm * 32 + i * 16 + sub) * 40 + kh * 8];
      af[i].q[0] = *(const uint4*)p;
      af[i].q[1] = *(const uint4*)(p + 16);
    }
    #pragma unroll
    for (int j = 0; j < 4; j++) {
      const unsigned short* p = &sB[cur][(wn * 64 + j * 16 + sub) * 40 + kh * 16];
      bf4[j].q[0] = *(const uint4*)p;
      bf4[j].q[1] = *(const uint4*)(p + 8);
    }
    #pragma unroll
    for (int i = 0; i < 2; i++)
      #pragma unroll
      for (int j = 0; j < 4; j++)
        acc[i][j] = __builtin_amdgcn_wmma_f32_16x16x32_bf16(
            false, af[i].v, false, bf4[j].v, (short)0, acc[i][j], false, false);
    if (have) {
      *(uint4*)&sA[nxt][lrow * 40 + lcol]     = a0;
      *(uint4*)&sA[nxt][lrow * 40 + lcol + 8] = a1;
      *(uint4*)&sB[nxt][lrow * 40 + lcol]     = b0;
      *(uint4*)&sB[nxt][lrow * 40 + lcol + 8] = b1;
    }
    __syncthreads();
  }
#endif

  // epilogue: elem r of lane -> M = r + kh*8 (consecutive r), N = sub
  #pragma unroll
  for (int i = 0; i < 2; i++) {
    #pragma unroll
    for (int j = 0; j < 4; j++) {
      int gm = m0 + wm * 32 + i * 16 + kh * 8;   // rows gm..gm+7
      int gn = n0 + wn * 64 + j * 16 + sub;
      float bv = bptr ? bptr[gn] : 0.0f;
      if (MODE == 0) {
        float* C = (float*)Cv + cOff;
        #pragma unroll
        for (int r = 0; r < 8; r++)
          C[(long)(gm + r) * ldc + gn] = acc[i][j][r] * scale + bv;
      } else if (MODE == 1) {
        unsigned short* C = (unsigned short*)Cv + cOff;
        #pragma unroll
        for (int r = 0; r < 8; r++) {
          float v = acc[i][j][r] + bv;
          if (RELU) v = v > 0.f ? v : 0.f;
          C[(long)(gm + r) * ldc + gn] = f2bf(v);
        }
      } else {  // MODE 2: transposed bf16 store out[b][gn][s], s = gm % rowsPerBatch
        unsigned short* C = (unsigned short*)Cv + cOff;
        int bb = gm / rowsPerBatch;
        int ss = gm % rowsPerBatch;
        union { unsigned short s[8]; uint4 q; } u;
        #pragma unroll
        for (int r = 0; r < 8; r++) u.s[r] = f2bf(acc[i][j][r] + bv);
        *(uint4*)&C[(long)bb * (long)N * rowsPerBatch + (long)gn * rowsPerBatch + ss] = u.q;
      }
    }
  }
}

// ---------------------------------------------------------------------------
// row softmax over 512 f32 elems -> bf16 (one block per row)
// ---------------------------------------------------------------------------
__global__ __launch_bounds__(256)
void softmax_rows512(const float* __restrict__ S, unsigned short* __restrict__ P) {
  long row = blockIdx.x;
  const float* x = S + row * 512;
  unsigned short* y = P + row * 512;
  int tid = threadIdx.x;
  float a = x[tid], b = x[tid + 256];
  __shared__ float red[256];
  red[tid] = fmaxf(a, b);
  __syncthreads();
  for (int s = 128; s > 0; s >>= 1) {
    if (tid < s) red[tid] = fmaxf(red[tid], red[tid + s]);
    __syncthreads();
  }
  float m = red[0];
  __syncthreads();
  float ea = __expf(a - m), eb = __expf(b - m);
  red[tid] = ea + eb;
  __syncthreads();
  for (int s = 128; s > 0; s >>= 1) {
    if (tid < s) red[tid] += red[tid + s];
    __syncthreads();
  }
  float inv = 1.0f / red[0];
  y[tid]       = f2bf(ea * inv);
  y[tid + 256] = f2bf(eb * inv);
}

// ---------------------------------------------------------------------------
// layernorm over (S*H) per batch: stats (mean, rstd) then apply
// ---------------------------------------------------------------------------
__global__ __launch_bounds__(256)
void ln_stats(const float* __restrict__ x1, const float* __restrict__ x2,
              float* __restrict__ stats, long perB) {
  int b = blockIdx.x;
  const float* p1 = x1 + (long)b * perB;
  const float* p2 = x2 + (long)b * perB;
  float s = 0.f, ss = 0.f;
  for (long i = threadIdx.x; i < perB; i += 256) {
    float v = p1[i] + p2[i];
    s += v; ss += v * v;
  }
  __shared__ float rs[256], rq[256];
  rs[threadIdx.x] = s; rq[threadIdx.x] = ss;
  __syncthreads();
  for (int t = 128; t > 0; t >>= 1) {
    if ((int)threadIdx.x < t) { rs[threadIdx.x] += rs[threadIdx.x + t]; rq[threadIdx.x] += rq[threadIdx.x + t]; }
    __syncthreads();
  }
  if (threadIdx.x == 0) {
    float mu  = rs[0] / (float)perB;
    float var = rq[0] / (float)perB - mu * mu;
    stats[b * 2]     = mu;
    stats[b * 2 + 1] = rsqrtf(var + 1e-5f);
  }
}

__global__ void ln_apply(const float* __restrict__ x1, const float* __restrict__ x2,
                         const float* __restrict__ stats,
                         float* __restrict__ outF, unsigned short* __restrict__ outB,
                         long perB) {
  int b = blockIdx.y;
  float mu = stats[b * 2], r = stats[b * 2 + 1];
  long base = (long)b * perB;
  for (long i = (long)blockIdx.x * blockDim.x + threadIdx.x; i < perB;
       i += (long)gridDim.x * blockDim.x) {
    float v = (x1[base + i] + x2[base + i] - mu) * r;
    if (outF) outF[base + i] = v;
    if (outB) outB[base + i] = f2bf(v);
  }
}

// ---------------------------------------------------------------------------
extern "C" void kernel_launch(void* const* d_in, const int* in_sizes, int n_in,
                              void* d_out, int out_size, void* d_ws, size_t ws_size,
                              hipStream_t stream) {
  (void)in_sizes; (void)n_in; (void)out_size; (void)ws_size;
  const float* state = (const float*)d_in[0];
  const float* Wq = (const float*)d_in[1];
  const float* bq = (const float*)d_in[2];
  const float* Wk = (const float*)d_in[3];
  const float* bk = (const float*)d_in[4];
  const float* Wv = (const float*)d_in[5];
  const float* bv = (const float*)d_in[6];
  const float* Wo = (const float*)d_in[7];
  const float* bo = (const float*)d_in[8];
  const float* W1 = (const float*)d_in[9];
  const float* b1 = (const float*)d_in[10];
  const float* W2 = (const float*)d_in[11];
  const float* b2 = (const float*)d_in[12];

  const long NB = 16, S = 512, H = 768, NH = 12, M = NB * S;  // M = 8192

  size_t off = 0;
  auto alloc = [&](size_t bytes) {
    char* p = (char*)d_ws + off;
    off += (bytes + 255) & ~(size_t)255;
    return (void*)p;
  };
  unsigned short* stateB = (unsigned short*)alloc(M * H * 2);
  unsigned short* WqT    = (unsigned short*)alloc(NH * H * H * 2);
  unsigned short* WkT    = (unsigned short*)alloc(NH * H * H * 2);
  unsigned short* WvT    = (unsigned short*)alloc(NH * H * H * 2);
  unsigned short* WoT    = (unsigned short*)alloc(NH * H * H * 2);
  unsigned short* W1T    = (unsigned short*)alloc(H * H * 2);
  unsigned short* W2T    = (unsigned short*)alloc(H * H * 2);
  unsigned short* Qb     = (unsigned short*)alloc(NH * M * H * 2);
  unsigned short* Kb     = (unsigned short*)alloc(NH * M * H * 2);
  unsigned short* Vt     = (unsigned short*)alloc(NH * NB * H * S * 2);
  float*          scoresF= (float*)alloc(NH * NB * S * S * 4);
  unsigned short* Pb     = (unsigned short*)alloc(NH * NB * S * S * 2);
  unsigned short* cat    = (unsigned short*)alloc(M * NH * H * 2);
  float*          attnF  = (float*)alloc(M * H * 4);
  float*          norm1F = (float*)alloc(M * H * 4);
  unsigned short* norm1B = (unsigned short*)alloc(M * H * 2);
  unsigned short* h1     = (unsigned short*)alloc(M * H * 2);
  float*          hidF   = (float*)alloc(M * H * 4);
  float*          stats1 = (float*)alloc(NB * 2 * 4);
  float*          stats2 = (float*)alloc(NB * 2 * 4);

  // 1) convert state, transpose-convert weights to bf16 [N,K]
  convert_f32_bf16<<<dim3((unsigned)((M * H / 4 + 255) / 256)), 256, 0, stream>>>(state, stateB, M * H);
  dim3 tb(32, 8);
  transpose_f32_bf16<<<dim3(24, 24, 12), tb, 0, stream>>>(Wq, WqT, 768, 768);
  transpose_f32_bf16<<<dim3(24, 24, 12), tb, 0, stream>>>(Wk, WkT, 768, 768);
  transpose_f32_bf16<<<dim3(24, 24, 12), tb, 0, stream>>>(Wv, WvT, 768, 768);
  transpose_f32_bf16<<<dim3(24, 288, 1), tb, 0, stream>>>(Wo, WoT, 9216, 768);
  transpose_f32_bf16<<<dim3(24, 24, 1),  tb, 0, stream>>>(W1, W1T, 768, 768);
  transpose_f32_bf16<<<dim3(24, 24, 1),  tb, 0, stream>>>(W2, W2T, 768, 768);

  // 2) per-head projections: Q,K bf16 [h][8192][768]; V stored transposed [h][b][768][512]
  gemm_wmma<1,false><<<dim3(6, 64, 12), 256, 0, stream>>>(
      stateB, WqT, Qb, bq, 8192, 768, 768, 768, 768, 768, 1,
      0, 0, 768L*768, 0, 8192L*768, 0, 768, 0, 1.0f, 0);
  gemm_wmma<1,false><<<dim3(6, 64, 12), 256, 0, stream>>>(
      stateB, WkT, Kb, bk, 8192, 768, 768, 768, 768, 768, 1,
      0, 0, 768L*768, 0, 8192L*768, 0, 768, 0, 1.0f, 0);
  gemm_wmma<2,false><<<dim3(6, 64, 12), 256, 0, stream>>>(
      stateB, WvT, Vt, bv, 8192, 768, 768, 768, 768, 512, 1,
      0, 0, 768L*768, 0, 16L*768*512, 0, 768, 0, 1.0f, 512);

  // 3) scores[h,b] = Q Kt / sqrt(768)  (z = h*16+b; A/B stride 512*768 works for both)
  const float scaleQK = 1.0f / sqrtf(768.0f);
  gemm_wmma<0,false><<<dim3(4, 4, 192), 256, 0, stream>>>(
      Qb, Kb, scoresF, nullptr, 512, 512, 768, 768, 768, 512, 1,
      512L*768, 0, 512L*768, 0, 512L*512, 0, 0, 0, scaleQK, 0);

  // 4) softmax rows -> bf16 P
  softmax_rows512<<<dim3(192 * 512), 256, 0, stream>>>(scoresF, Pb);

  // 5) heads = P @ V, written directly into cat[b*512+s][h*768+e]
  gemm_wmma<1,false><<<dim3(6, 4, 192), 256, 0, stream>>>(
      Pb, Vt, cat, nullptr, 512, 768, 512, 512, 512, 9216, 16,
      16L*512*512, 512L*512, 16L*768*512, 768L*512, 768, 512L*9216, 0, 0, 1.0f, 0);

  // 6) attention = cat @ Wo + bo  (f32)
  gemm_wmma<0,false><<<dim3(6, 64, 1), 256, 0, stream>>>(
      cat, WoT, attnF, bo, 8192, 768, 9216, 9216, 9216, 768, 1,
      0, 0, 0, 0, 0, 0, 0, 0, 1.0f, 0);

  // 7) norm1 = LN(state + attention) over (S,H) per batch
  ln_stats<<<dim3(16), 256, 0, stream>>>(state, attnF, stats1, S * H);
  ln_apply<<<dim3(256, 16), 256, 0, stream>>>(state, attnF, stats1, norm1F, norm1B, S * H);

  // 8) FFN: relu(norm1 @ W1 + b1) @ W2 + b2
  gemm_wmma<1,true><<<dim3(6, 64, 1), 256, 0, stream>>>(
      norm1B, W1T, h1, b1, 8192, 768, 768, 768, 768, 768, 1,
      0, 0, 0, 0, 0, 0, 0, 0, 1.0f, 0);
  gemm_wmma<0,false><<<dim3(6, 64, 1), 256, 0, stream>>>(
      h1, W2T, hidF, b2, 8192, 768, 768, 768, 768, 768, 1,
      0, 0, 0, 0, 0, 0, 0, 0, 1.0f, 0);

  // 9) out = LN(norm1 + hidden)
  ln_stats<<<dim3(16), 256, 0, stream>>>(norm1F, hidF, stats2, S * H);
  ln_apply<<<dim3(256, 16), 256, 0, stream>>>(norm1F, hidF, stats2, (float*)d_out, nullptr, S * H);
}
